// SparseMHA_11991548691188
// MI455X (gfx1250) — matrix operand order repaired and checked by reading, not compiled, not run on hardware
//
#include <hip/hip_runtime.h>

// ---------------------------------------------------------------------------
// Sparse multi-head graph attention for MI455X (gfx1250, wave32).
//   N=50000 nodes, E=1.6M edges, HIDDEN=128, 8 heads x 16 dims.
// Pipeline:
//   1) h -> bf16 copy; W{q,k,v} -> bf16 pre-swizzled into WMMA B-fragment order
//   2) fused QKV projection: per block, async-stage the 96KB weight-fragment
//      table into LDS (global_load_async_to_lds_b128 + s_wait_asynccnt), then
//      4 waves x 1 M-tile x 24 N-tiles of v_wmma_f32_16x16x32_bf16 (f32 accum),
//      bias + q-scaling folded in, stored head-deinterleaved: [node][head*16+d]
//   3) SDDMM per (edge, head): dot16 + atomic segment-max (ordered-uint trick)
//   4) exp(score - rowmax) + atomic segment-sum
//   5) SpMM scatter: atomicAdd of w * v into per-node accumulator
//   6) normalize by rowsum and permute into reference layout [node][d*8+h]
// ---------------------------------------------------------------------------

#define HID 128
#define NH  8
#define HD  16
#define QSCALE 0.25f  // HEAD_DIM^-0.5 = 16^-0.5
#define WF_ELEMS (3 * 8 * 4 * 32 * 16)   // 49152 bf16 = 96 KB

typedef __attribute__((ext_vector_type(16))) __bf16 v16bf;
typedef __attribute__((ext_vector_type(8)))  __bf16 v8bf;
typedef __attribute__((ext_vector_type(8)))  float  v8f;

static __device__ __forceinline__ __bf16 f2bf(float f) { return (__bf16)f; }

// monotone encoding of float into uint so unsigned atomicMax == float max
static __device__ __forceinline__ unsigned f2ord(float f) {
  unsigned u = __float_as_uint(f);
  return (u & 0x80000000u) ? ~u : (u | 0x80000000u);
}
static __device__ __forceinline__ float ord2f(unsigned u) {
  unsigned v = (u & 0x80000000u) ? (u ^ 0x80000000u) : ~u;
  return __uint_as_float(v);
}
#define ORD_NEG_INF 0x007FFFFFu  // f2ord(-inf)

// --------------------------- setup kernels ---------------------------------

__global__ void k_convert_h(const float* __restrict__ h, __bf16* __restrict__ hbf,
                            int total4) {
  int i = blockIdx.x * blockDim.x + threadIdx.x;
  if (i >= total4) return;
  const float4 f = ((const float4*)h)[i];
  __bf16* o = hbf + (size_t)i * 4;
  o[0] = f2bf(f.x); o[1] = f2bf(f.y); o[2] = f2bf(f.z); o[3] = f2bf(f.w);
}

// Pre-swizzle W into WMMA bf16 B-fragment order.
// B matrix is 32(K) x 16(N): lane L holds column N=L%16, K = (L/16)*8 + (j&7) + (j>>3)*16.
// wfrag index: (((m*8 + t)*4 + s)*32 + L)*16 + j   (one 32B chunk per lane)
__global__ void k_build_wfrag(const float* __restrict__ Wq,
                              const float* __restrict__ Wk,
                              const float* __restrict__ Wv,
                              __bf16* __restrict__ wfrag) {
  int idx = blockIdx.x * blockDim.x + threadIdx.x;
  if (idx >= WF_ELEMS) return;
  int j = idx & 15;
  int L = (idx >> 4) & 31;
  int s = (idx >> 9) & 3;
  int t = (idx >> 11) & 7;
  int m = idx >> 14;
  int K = s * 32 + (L >> 4) * 8 + (j & 7) + (j >> 3) * 16;     // input-dim row of W
  int Nc = t * 16 + (L & 15);             // output column
  const float* W = (m == 0) ? Wq : (m == 1) ? Wk : Wv;
  wfrag[idx] = f2bf(W[(size_t)K * HID + Nc]);
}

__global__ void k_init(float* __restrict__ acc, unsigned* __restrict__ rmax,
                       float* __restrict__ rsum, int n) {
  int i = blockIdx.x * blockDim.x + threadIdx.x;
  if (i < n * HID) acc[i] = 0.0f;
  if (i < n * NH) { rmax[i] = ORD_NEG_INF; rsum[i] = 0.0f; }
}

// --------------------------- QKV projection (WMMA) -------------------------
// Block = 128 threads = 4 waves, covering 4 M-tiles (64 rows).
// The 96KB pre-swizzled weight-fragment table is async-copied global->LDS once
// per block (ASYNCcnt-tracked), then each wave runs all 24 output tiles
// (3 matrices x 8 N-tiles) against its own A-fragments -> 96 WMMAs/wave.
__global__ void __launch_bounds__(128)
k_qkv_proj(const __bf16* __restrict__ hbf, const __bf16* __restrict__ wfrag,
           const float* __restrict__ bq, const float* __restrict__ bk,
           const float* __restrict__ bv, float* __restrict__ qs,
           float* __restrict__ ks, float* __restrict__ vs, int n) {
  __shared__ __align__(64) __bf16 swf[WF_ELEMS];
  const int tid = threadIdx.x;

  // ---- async stage: 98304B = 6144 x b128, 48 chunks per thread ----
  {
    unsigned ldsbase = (unsigned)(unsigned long long)(&swf[0]);
#pragma unroll
    for (int i = 0; i < 48; ++i) {
      unsigned off = (unsigned)(tid + i * 128) * 16u;
      unsigned lds = ldsbase + off;
      asm volatile("global_load_async_to_lds_b128 %0, %1, %2"
                   :: "v"(lds), "v"(off), "s"(wfrag) : "memory");
    }
    asm volatile("s_wait_asynccnt 0" ::: "memory");
  }
  __syncthreads();

  const int wave = tid >> 5;
  const int L = tid & 31;
  const int g = L >> 4;                    // lane half (K group)
  const int mtile = blockIdx.x * 4 + wave;
  if (mtile * 16 >= n) return;             // wave-uniform exit (EXEC stays full)
  const bool full = (mtile * 16 + 16) <= n;

  int Mrow = mtile * 16 + (L & 15);
  int MrowLd = Mrow < n ? Mrow : (n - 1);

  // A-fragments: 16-bit A 16x32 layout — lane holds K in {g*8..g*8+7} and
  // {16+g*8..16+g*8+7}; two contiguous 16B runs per lane per k-step.
  v16bf a[4];
#pragma unroll
  for (int s = 0; s < 4; ++s) {
    const __bf16* p = hbf + (size_t)MrowLd * HID + s * 32 + g * 8;
    v8bf lo = *(const v8bf*)p;
    v8bf hi = *(const v8bf*)(p + 16);
    a[s] = __builtin_shufflevector(lo, hi, 0, 1, 2, 3, 4, 5, 6, 7,
                                   8, 9, 10, 11, 12, 13, 14, 15);
  }

  for (int m = 0; m < 3; ++m) {
    const float* bvec = (m == 0) ? bq : (m == 1) ? bk : bv;
    const float scale = (m == 0) ? QSCALE : 1.0f;
    float* dst = (m == 0) ? qs : (m == 1) ? ks : vs;
#pragma unroll
    for (int t = 0; t < 8; ++t) {
      v8f c = {};
      const __bf16* wf = swf + ((size_t)((m * 8 + t) * 4) * 32 + L) * 16;
#pragma unroll
      for (int s = 0; s < 4; ++s) {
        v16bf b = *(const v16bf*)(wf + (size_t)s * 512);  // ds_load_b128 x2
        c = __builtin_amdgcn_wmma_f32_16x16x32_bf16(
            false, a[s], false, b, (short)0, c, false, false);
      }
      const int ccol = t * 16 + (L & 15);
      const float bias = bvec[ccol];
      // de-interleave: proj column c = d*8+h  ->  ws column h*16+d
      const int wcol = (ccol & 7) * 16 + (ccol >> 3);
      float* base = dst + (size_t)(mtile * 16 + g * 8) * HID + wcol;
      if (full) {
#pragma unroll
        for (int r = 0; r < 8; ++r) base[(size_t)r * HID] = (c[r] + bias) * scale;
      } else {
#pragma unroll
        for (int r = 0; r < 8; ++r) {
          int row = mtile * 16 + r + g * 8;
          if (row < n) dst[(size_t)row * HID + wcol] = (c[r] + bias) * scale;
        }
      }
    }
  }
}

// --------------------------- edge kernels ----------------------------------
// One thread per (edge, head); lanes map heads fastest -> 512B coalesced
// gathers per 8-lane cluster, all hitting the 192MB L2 (q/k/v = 76.8MB).

__global__ void k_sddmm(const int* __restrict__ erow, const int* __restrict__ ecol,
                        const float* __restrict__ qs, const float* __restrict__ ks,
                        float* __restrict__ scores, unsigned* __restrict__ rmax,
                        int nE) {
  long long idx = (long long)blockIdx.x * blockDim.x + threadIdx.x;
  if (idx >= (long long)nE * NH) return;
  int e = (int)(idx >> 3), hh = (int)(idx & 7);
  __builtin_prefetch(erow + e + 2048, 0, 0);  // stream edge list ahead
  int r = erow[e], c = ecol[e];
  const float4* qp = (const float4*)(qs + (size_t)r * HID + hh * HD);
  const float4* kp = (const float4*)(ks + (size_t)c * HID + hh * HD);
  float dot = 0.0f;
#pragma unroll
  for (int i = 0; i < 4; ++i) {
    float4 aq = qp[i], bk4 = kp[i];
    dot += aq.x * bk4.x + aq.y * bk4.y + aq.z * bk4.z + aq.w * bk4.w;
  }
  scores[idx] = dot;
  atomicMax(rmax + (size_t)r * NH + hh, f2ord(dot));
}

__global__ void k_softmax_exp(const int* __restrict__ erow,
                              float* __restrict__ scores,
                              const unsigned* __restrict__ rmax,
                              float* __restrict__ rsum, int nE) {
  long long idx = (long long)blockIdx.x * blockDim.x + threadIdx.x;
  if (idx >= (long long)nE * NH) return;
  int e = (int)(idx >> 3), hh = (int)(idx & 7);
  int r = erow[e];
  float m = ord2f(rmax[(size_t)r * NH + hh]);
  float ex = __expf(scores[idx] - m);
  scores[idx] = ex;                       // in-place: scores -> unnormalized attn
  atomicAdd(rsum + (size_t)r * NH + hh, ex);
}

__global__ void k_spmm(const int* __restrict__ erow, const int* __restrict__ ecol,
                       const float* __restrict__ scores,
                       const float* __restrict__ vs, float* __restrict__ acc,
                       int nE) {
  long long idx = (long long)blockIdx.x * blockDim.x + threadIdx.x;
  if (idx >= (long long)nE * NH) return;
  int e = (int)(idx >> 3), hh = (int)(idx & 7);
  __builtin_prefetch(ecol + e + 2048, 0, 0);
  int r = erow[e], c = ecol[e];
  float w = scores[idx];
  const float4* vp = (const float4*)(vs + (size_t)c * HID + hh * HD);
  float* ap = acc + (size_t)r * HID + hh * HD;
#pragma unroll
  for (int i = 0; i < 4; ++i) {
    float4 v4 = vp[i];
    atomicAdd(ap + i * 4 + 0, w * v4.x);
    atomicAdd(ap + i * 4 + 1, w * v4.y);
    atomicAdd(ap + i * 4 + 2, w * v4.z);
    atomicAdd(ap + i * 4 + 3, w * v4.w);
  }
}

__global__ void k_normalize(const float* __restrict__ acc,
                            const float* __restrict__ rsum,
                            float* __restrict__ out, int n) {
  int idx = blockIdx.x * blockDim.x + threadIdx.x;
  if (idx >= n * HID) return;
  int i = idx >> 7, cc = idx & 127;
  int hh = cc & 7, d = cc >> 3;           // reference layout: [i][d*8+h]
  out[idx] = acc[(size_t)i * HID + hh * HD + d] / rsum[(size_t)i * NH + hh];
}

// --------------------------- launcher --------------------------------------

extern "C" void kernel_launch(void* const* d_in, const int* in_sizes, int n_in,
                              void* d_out, int out_size, void* d_ws, size_t ws_size,
                              hipStream_t stream) {
  const float* h    = (const float*)d_in[0];
  const int*   erow = (const int*)d_in[1];
  const int*   ecol = (const int*)d_in[2];
  const float* Wq   = (const float*)d_in[3];
  const float* bq   = (const float*)d_in[4];
  const float* Wk   = (const float*)d_in[5];
  const float* bk   = (const float*)d_in[6];
  const float* Wv   = (const float*)d_in[7];
  const float* bv   = (const float*)d_in[8];
  float* out = (float*)d_out;

  const int n = in_sizes[0] / HID;   // 50000
  const int E = in_sizes[1];         // 1600000

  // workspace carve-up (all offsets 128B-aligned)
  char* ws = (char*)d_ws;
  size_t o = 0;
  float*   qs    = (float*)(ws + o);  o += (size_t)n * HID * 4;
  float*   kks   = (float*)(ws + o);  o += (size_t)n * HID * 4;
  float*   vvs   = (float*)(ws + o);  o += (size_t)n * HID * 4;
  __bf16*  hbf   = (__bf16*)(ws + o); o += (size_t)n * HID * 2;
  __bf16*  wfrag = (__bf16*)(ws + o); o += (size_t)WF_ELEMS * 2;
  float*   sc    = (float*)(ws + o);  o += (size_t)E * NH * 4;
  unsigned* rmax = (unsigned*)(ws + o); o += (size_t)n * NH * 4;
  float*   rsum  = (float*)(ws + o);  o += (size_t)n * NH * 4;
  float*   acc   = (float*)(ws + o);  o += (size_t)n * HID * 4;
  (void)ws_size; (void)n_in; (void)out_size;

  const int T = 256;
  // 1) precision conversion + weight swizzle
  int cvt4 = n * HID / 4;
  k_convert_h<<<(cvt4 + T - 1) / T, T, 0, stream>>>(h, hbf, cvt4);
  k_build_wfrag<<<(WF_ELEMS + T - 1) / T, T, 0, stream>>>(Wq, Wk, Wv, wfrag);
  k_init<<<(n * HID + T - 1) / T, T, 0, stream>>>(acc, rmax, rsum, n);

  // 2) fused QKV projection via WMMA (4 waves = 4 M-tiles per block,
  //    weight table async-staged in LDS)
  int mtiles = (n + 15) / 16;
  int pblocks = (mtiles + 3) / 4;
  k_qkv_proj<<<pblocks, 128, 0, stream>>>(hbf, wfrag, bq, bk, bv, qs, kks, vvs, n);

  // 3..5) edge phase
  long long eh = (long long)E * NH;
  int egrid = (int)((eh + T - 1) / T);
  k_sddmm<<<egrid, T, 0, stream>>>(erow, ecol, qs, kks, sc, rmax, E);
  k_softmax_exp<<<egrid, T, 0, stream>>>(erow, sc, rmax, rsum, E);
  k_spmm<<<egrid, T, 0, stream>>>(erow, ecol, sc, vvs, acc, E);

  // 6) normalize + permute to reference layout
  k_normalize<<<(n * HID + T - 1) / T, T, 0, stream>>>(acc, rsum, out, n);
}
